// WindowAttention_2525440770315
// MI455X (gfx1250) — compile-verified, matrix operand order checked by reading, not run on hardware
//
#include <hip/hip_runtime.h>

// ---------------------------------------------------------------------------
// Swin WindowAttention forward for MI455X (gfx1250), bf16 WMMA pipeline.
//   x[2048,49,512] f32 -> out[2048,49,512] f32
// Three kernels: qkv GEMM -> per-(window,head) attention -> proj GEMM.
// All matmuls use v_wmma_f32_16x16x32_bf16 (f32 accumulate).
// Staging and epilogues are 128-bit (b128) on both LDS and global paths.
// ---------------------------------------------------------------------------

typedef __attribute__((ext_vector_type(16))) __bf16        v16bf;
typedef __attribute__((ext_vector_type(8)))  float         v8f;
typedef __attribute__((ext_vector_type(4)))  float         f32x4;
typedef __attribute__((ext_vector_type(4)))  unsigned int  u32x4;

#define WIN_N  49
#define NPAD   64
#define DIM    512
#define HEADS  16
#define HD     32
#define BWIN   2048
#define NWMASK 64

// 16-bit A/B operand K packing (CDNA5 ISA 7.12.2, 16-bit A-matrix 16x32):
// lane-half h, operand VGPR v holds the K pair starting at:
__device__ __forceinline__ int kbase(int v, int half) {
    return ((v < 4) ? 0 : 16) + half * 8 + (v & 3) * 2;
}

union BFVec  { v16bf v; unsigned int u[8]; };   // one WMMA 16-bit operand
union Pack8  { __bf16 h[8]; u32x4 q; };         // 8 bf16 <-> one b128
union Pack2  { __bf16 h[2]; unsigned int u; };  // 2 bf16 <-> one dword

__device__ __forceinline__ v8f wmma_bf16(const BFVec& a, const BFVec& b, v8f c) {
    // (neg_a, A, neg_b, B, c_mod, C, reuse_a, reuse_b)
    return __builtin_amdgcn_wmma_f32_16x16x32_bf16(false, a.v, false, b.v,
                                                   (short)0, c, false, false);
}

__device__ __forceinline__ void pack8_f32(Pack8& t, const f32x4 f0, const f32x4 f1) {
    t.h[0] = (__bf16)f0.x; t.h[1] = (__bf16)f0.y;
    t.h[2] = (__bf16)f0.z; t.h[3] = (__bf16)f0.w;
    t.h[4] = (__bf16)f1.x; t.h[5] = (__bf16)f1.y;
    t.h[6] = (__bf16)f1.z; t.h[7] = (__bf16)f1.w;
}

#define V8F_ZERO {0.f,0.f,0.f,0.f,0.f,0.f,0.f,0.f}

// ---------------------------------------------------------------------------
// Kernel 1: qkv = x @ w_qkv^T + b_qkv -> Q, K ([tok][d]) and V^T ([d][tok]),
// bf16, tokens padded 49->64, q pre-scaled by 1/sqrt(hd).
// grid (2048 windows, 6 column-blocks of 256), block 256 (8 waves).
// Wave w owns a 64x32 tile == exactly one (q|k|v, head) slice.
// ---------------------------------------------------------------------------
__global__ __launch_bounds__(256)
void qkv_gemm_kernel(const float* __restrict__ x,
                     const float* __restrict__ w_qkv,
                     const float* __restrict__ b_qkv,
                     __bf16* __restrict__ Qb,
                     __bf16* __restrict__ Kb,
                     __bf16* __restrict__ Vtb)
{
    extern __shared__ __align__(16) char smem[];
    __bf16* As = (__bf16*)smem;                 // [64][32] bf16
    __bf16* Bs = (__bf16*)(smem + 64*32*2);     // [256][32] bf16, [col][k]

    const int b        = blockIdx.x;
    const int colBlock = blockIdx.y * 256;
    const int tid  = threadIdx.x;
    const int wave = tid >> 5, lane = tid & 31;
    const int half = lane >> 4, lc = lane & 15;

    v8f acc[4][2] = { { V8F_ZERO, V8F_ZERO }, { V8F_ZERO, V8F_ZERO },
                      { V8F_ZERO, V8F_ZERO }, { V8F_ZERO, V8F_ZERO } };

    const int rowA = tid >> 2;           // 0..63 (4 threads per row, 8 elems each)
    const int kkA  = (tid & 3) * 8;
    const float* xrow = x + ((size_t)b * WIN_N + rowA) * DIM;   // deref guarded
    const float* wrow = w_qkv + (size_t)(colBlock + tid) * DIM; // one col/thread

    for (int kb = 0; kb < DIM; kb += 32) {
        // stage A tile: one branch, b128 loads, packed bf16, one b128 LDS store
        Pack8 ta;
        if (rowA < WIN_N) {
            pack8_f32(ta, *(const f32x4*)(xrow + kb + kkA),
                          *(const f32x4*)(xrow + kb + kkA + 4));
        } else {
            ta.q = (u32x4){0u, 0u, 0u, 0u};
        }
        *(u32x4*)(As + rowA * 32 + kkA) = ta.q;

        // stage B tile transposed to [col][k]; b128 loads + b128 LDS stores
        #pragma unroll
        for (int g = 0; g < 4; g++) {
            Pack8 tb;
            pack8_f32(tb, *(const f32x4*)(wrow + kb + g*8),
                          *(const f32x4*)(wrow + kb + g*8 + 4));
            *(u32x4*)(Bs + tid * 32 + g*8) = tb.q;
        }
        // keep next weight tile flowing into L2 (global_prefetch_b8)
        __builtin_prefetch((const void*)(wrow + kb + 32), 0, 1);
        __syncthreads();

        BFVec a[4], bb;
        #pragma unroll
        for (int mt = 0; mt < 4; mt++) {
            const __bf16* ap = As + (mt*16 + lc) * 32;
            #pragma unroll
            for (int v = 0; v < 8; v++)
                a[mt].u[v] = *(const unsigned int*)(ap + kbase(v, half));
        }
        #pragma unroll
        for (int nt = 0; nt < 2; nt++) {
            const __bf16* bp = Bs + (wave*32 + nt*16 + lc) * 32;
            #pragma unroll
            for (int v = 0; v < 8; v++)
                bb.u[v] = *(const unsigned int*)(bp + kbase(v, half));
            #pragma unroll
            for (int mt = 0; mt < 4; mt++)
                acc[mt][nt] = wmma_bf16(a[mt], bb, acc[mt][nt]);
        }
        __syncthreads();
    }

    // ---- epilogue: wave-uniform routing (tile = one (part, head) slice) ----
    const int   obase = colBlock + wave * 32;        // 32-aligned
    const int   part  = obase >> 9;                  // 0=q 1=k 2=v (wave-uniform)
    const int   h     = (obase >> 5) & 15;           // head (wave-uniform)
    const float smul  = (part == 0) ? 0.17677669529663689f : 1.0f;
    const float bias0 = b_qkv[obase + lc];
    const float bias1 = b_qkv[obase + 16 + lc];
    const size_t bh   = (size_t)b * HEADS + h;

    // per-wave 4KB LDS staging region (smem reusable after final barrier)
    __bf16* Ls = (__bf16*)(smem + wave * 4096);
    #pragma unroll
    for (int mt = 0; mt < 4; mt++)
      #pragma unroll
      for (int nt = 0; nt < 2; nt++)
        #pragma unroll
        for (int r = 0; r < 8; r++) {
            int   m = mt*16 + r + half*8;            // padded token
            int   d = nt*16 + lc;
            float v = (acc[mt][nt][r] + (nt ? bias1 : bias0)) * smul;
            if (part < 2) Ls[m*32 + d] = (__bf16)v;  // [tok][d]
            else          Ls[d*64 + m] = (__bf16)v;  // [d][tok]
        }

    if (part < 2) {
        __bf16* dst = (part == 0 ? Qb : Kb) + bh * NPAD * HD;
        #pragma unroll
        for (int rr = 0; rr < 2; rr++) {             // 2 rows/lane, 64B each
            int m = lane + rr * 32;
            #pragma unroll
            for (int g = 0; g < 4; g++)
                *(u32x4*)(dst + m*32 + g*8) = *(const u32x4*)(Ls + m*32 + g*8);
        }
    } else {
        __bf16* dst = Vtb + bh * HD * NPAD;
        int d = lane;                                // 1 row/lane, 128B each
        #pragma unroll
        for (int g = 0; g < 8; g++)
            *(u32x4*)(dst + d*64 + g*8) = *(const u32x4*)(Ls + d*64 + g*8);
    }
}

// ---------------------------------------------------------------------------
// Kernel 2: per (window, head) attention. 1 wave per head, 4 heads/block.
// S = Q K^T (64x64 pad) -> +rpb +mask -> softmax -> P (bf16, pad cols = 0)
// O = P V -> Y[b, tok, h*32+d] bf16 (b128 stores via LDS restage).
// grid (2048, 4), block 128.
// ---------------------------------------------------------------------------
__global__ __launch_bounds__(128)
void attn_kernel(const __bf16* __restrict__ Qb,
                 const __bf16* __restrict__ Kb,
                 const __bf16* __restrict__ Vtb,
                 const float*  __restrict__ bias_table,
                 const int*    __restrict__ rel_index,
                 const float*  __restrict__ mask,
                 __bf16* __restrict__ Y)
{
    extern __shared__ __align__(16) char smem[];
    const int b    = blockIdx.x;
    const int tid  = threadIdx.x;
    const int wave = tid >> 5, lane = tid & 31;
    const int half = lane >> 4, lc = lane & 15;
    const int h    = blockIdx.y * 4 + wave;

    // per-wave private LDS region: 64x64 f32 scores + 64x64 bf16 probs
    float*  Sf = (float*)(smem + (size_t)wave * (64*64*4 + 64*64*2));
    __bf16* P  = (__bf16*)((char*)Sf + 64*64*4);

    const size_t bh = (size_t)b * HEADS + h;
    const __bf16* Qh  = Qb  + bh * NPAD * HD;   // [tok][d]
    const __bf16* Kh  = Kb  + bh * NPAD * HD;   // [tok][d]
    const __bf16* Vth = Vtb + bh * HD * NPAD;   // [d][tok]

    // ---- Phase 1: S = Q K^T, K-dim = 32 (single wmma per 16x16 tile)
    #pragma unroll
    for (int mt = 0; mt < 4; mt++) {
        BFVec a;
        const __bf16* ap = Qh + (mt*16 + lc) * HD;
        #pragma unroll
        for (int v = 0; v < 8; v++)
            a.u[v] = *(const unsigned int*)(ap + kbase(v, half));
        #pragma unroll
        for (int nt = 0; nt < 4; nt++) {
            BFVec bb;   // B = K^T: lane = key token, K-dim = d (contiguous pairs)
            const __bf16* bp = Kh + (nt*16 + lc) * HD;
            #pragma unroll
            for (int v = 0; v < 8; v++)
                bb.u[v] = *(const unsigned int*)(bp + kbase(v, half));
            v8f c = V8F_ZERO;
            c = wmma_bf16(a, bb, c);
            #pragma unroll
            for (int r = 0; r < 8; r++)
                Sf[(mt*16 + r + half*8) * 64 + nt*16 + lc] = c[r];
        }
    }
    __syncthreads();

    // ---- Phase 2: softmax with rel-pos bias + shift mask (ref: -100 adds)
    const float* mrow = mask + (size_t)(b & (NWMASK-1)) * WIN_N * WIN_N;
    const u32x4 z4 = {0u, 0u, 0u, 0u};
    for (int rr = 0; rr < 2; rr++) {
        int m = lane + rr * 32;
        if (m < WIN_N) {
            float mx = -3.4e38f;
            for (int j = 0; j < WIN_N; j++) {
                float t = Sf[m*64 + j]
                        + bias_table[rel_index[m*WIN_N + j] * HEADS + h]
                        + mrow[m*WIN_N + j];
                Sf[m*64 + j] = t;
                mx = fmaxf(mx, t);
            }
            float sum = 0.f;
            for (int j = 0; j < WIN_N; j++) {
                float e = __expf(Sf[m*64 + j] - mx);
                Sf[m*64 + j] = e;
                sum += e;
            }
            float inv = 1.f / sum;
            for (int j = 0; j < 64; j += 2) {        // dword-packed P writes
                Pack2 p;
                p.h[0] = (__bf16)((j     < WIN_N) ? Sf[m*64 + j    ] * inv : 0.f);
                p.h[1] = (__bf16)((j + 1 < WIN_N) ? Sf[m*64 + j + 1] * inv : 0.f);
                *(unsigned int*)(P + m*64 + j) = p.u;
            }
        } else {
            #pragma unroll
            for (int g = 0; g < 8; g++)              // zero pad-query rows, b128
                *(u32x4*)(P + m*64 + g*8) = z4;
        }
    }
    __syncthreads();

    // ---- Phase 3: O = P V, K-dim = 64 tokens (two wmma k-steps)
    __bf16* Ob = (__bf16*)Sf;                        // reuse score region, [tok][d]
    #pragma unroll
    for (int mt = 0; mt < 4; mt++) {
        #pragma unroll
        for (int nt = 0; nt < 2; nt++) {
            v8f c = V8F_ZERO;
            #pragma unroll
            for (int kk = 0; kk < 2; kk++) {
                BFVec a, bb;
                const __bf16* ap = P   + (mt*16 + lc) * 64   + kk*32; // [q][kv]
                const __bf16* bp = Vth + (nt*16 + lc) * NPAD + kk*32; // [d][kv]
                #pragma unroll
                for (int v = 0; v < 8; v++) {
                    a.u[v]  = *(const unsigned int*)(ap + kbase(v, half));
                    bb.u[v] = *(const unsigned int*)(bp + kbase(v, half));
                }
                c = wmma_bf16(a, bb, c);
            }
            #pragma unroll
            for (int r = 0; r < 8; r++)
                Ob[(mt*16 + r + half*8) * 32 + nt*16 + lc] = (__bf16)c[r];
        }
    }
    // coalesced Y stores: 2 rows/lane, 64B per row
    #pragma unroll
    for (int rr = 0; rr < 2; rr++) {
        int m = lane + rr * 32;
        if (m < WIN_N) {
            __bf16* dst = Y + ((size_t)b * WIN_N + m) * DIM + h * HD;
            #pragma unroll
            for (int g = 0; g < 4; g++)
                *(u32x4*)(dst + g*8) = *(const u32x4*)(Ob + m*32 + g*8);
        }
    }
}

// ---------------------------------------------------------------------------
// Kernel 3: out = Y @ w_proj^T + b_proj  (f32 output, b128 stores via LDS).
// grid (2048, 2 column-blocks of 256), block 256 (8 waves).
// ---------------------------------------------------------------------------
__global__ __launch_bounds__(256)
void proj_gemm_kernel(const __bf16* __restrict__ Y,
                      const float*  __restrict__ w_proj,
                      const float*  __restrict__ b_proj,
                      float* __restrict__ out)
{
    extern __shared__ __align__(16) char smem[];
    __bf16* As = (__bf16*)smem;                 // [64][32]
    __bf16* Bs = (__bf16*)(smem + 64*32*2);     // [256][32]

    const int b        = blockIdx.x;
    const int colBlock = blockIdx.y * 256;
    const int tid  = threadIdx.x;
    const int wave = tid >> 5, lane = tid & 31;
    const int half = lane >> 4, lc = lane & 15;

    v8f acc[4][2] = { { V8F_ZERO, V8F_ZERO }, { V8F_ZERO, V8F_ZERO },
                      { V8F_ZERO, V8F_ZERO }, { V8F_ZERO, V8F_ZERO } };

    const int rowA = tid >> 2;
    const int kkA  = (tid & 3) * 8;
    const __bf16* yrow = Y + ((size_t)b * WIN_N + rowA) * DIM;
    const float*  wrow = w_proj + (size_t)(colBlock + tid) * DIM;

    for (int kb = 0; kb < DIM; kb += 32) {
        // A tile: already bf16, one 16B copy per thread; zero-pad rows >= 49
        u32x4 ta;
        if (rowA < WIN_N) ta = *(const u32x4*)(yrow + kb + kkA);
        else              ta = (u32x4){0u, 0u, 0u, 0u};
        *(u32x4*)(As + rowA*32 + kkA) = ta;

        #pragma unroll
        for (int g = 0; g < 4; g++) {
            Pack8 tb;
            pack8_f32(tb, *(const f32x4*)(wrow + kb + g*8),
                          *(const f32x4*)(wrow + kb + g*8 + 4));
            *(u32x4*)(Bs + tid * 32 + g*8) = tb.q;
        }
        __builtin_prefetch((const void*)(wrow + kb + 32), 0, 1);
        __syncthreads();

        BFVec a[4], bb;
        #pragma unroll
        for (int mt = 0; mt < 4; mt++) {
            const __bf16* ap = As + (mt*16 + lc) * 32;
            #pragma unroll
            for (int v = 0; v < 8; v++)
                a[mt].u[v] = *(const unsigned int*)(ap + kbase(v, half));
        }
        #pragma unroll
        for (int nt = 0; nt < 2; nt++) {
            const __bf16* bp = Bs + (wave*32 + nt*16 + lc) * 32;
            #pragma unroll
            for (int v = 0; v < 8; v++)
                bb.u[v] = *(const unsigned int*)(bp + kbase(v, half));
            #pragma unroll
            for (int mt = 0; mt < 4; mt++)
                acc[mt][nt] = wmma_bf16(a[mt], bb, acc[mt][nt]);
        }
        __syncthreads();
    }

    // ---- epilogue: stage f32 tile per wave in LDS, then b128 row stores ----
    const int obase = colBlock + wave * 32;
    const float bias0 = b_proj[obase + lc];
    const float bias1 = b_proj[obase + 16 + lc];
    float* Ls = (float*)(smem + wave * 8192);        // [64][32] f32 per wave
    #pragma unroll
    for (int mt = 0; mt < 4; mt++)
      #pragma unroll
      for (int nt = 0; nt < 2; nt++)
        #pragma unroll
        for (int r = 0; r < 8; r++)
            Ls[(mt*16 + r + half*8) * 32 + nt*16 + lc] =
                acc[mt][nt][r] + (nt ? bias1 : bias0);

    #pragma unroll
    for (int rr = 0; rr < 2; rr++) {                 // 2 rows/lane, 128B each
        int m = lane + rr * 32;
        if (m < WIN_N) {
            float* dst = out + ((size_t)b * WIN_N + m) * DIM + obase;
            #pragma unroll
            for (int g = 0; g < 8; g++)
                *(f32x4*)(dst + g*4) = *(const f32x4*)(Ls + m*32 + g*4);
        }
    }
}

// ---------------------------------------------------------------------------
extern "C" void kernel_launch(void* const* d_in, const int* in_sizes, int n_in,
                              void* d_out, int out_size, void* d_ws, size_t ws_size,
                              hipStream_t stream)
{
    const float* x          = (const float*)d_in[0];
    const float* w_qkv      = (const float*)d_in[1];
    const float* b_qkv      = (const float*)d_in[2];
    const float* w_proj     = (const float*)d_in[3];
    const float* b_proj     = (const float*)d_in[4];
    const float* bias_table = (const float*)d_in[5];
    const int*   rel_index  = (const int*)d_in[6];
    const float* mask       = (const float*)d_in[7];
    float* out = (float*)d_out;

    // workspace layout (bf16): Q | K | Vt | Y   (~482 MiB total)
    const size_t EQ = (size_t)BWIN * HEADS * NPAD * HD;   // 67,108,864 elems
    __bf16* Qb  = (__bf16*)d_ws;
    __bf16* Kb  = Qb + EQ;
    __bf16* Vtb = Kb + EQ;
    __bf16* Yb  = Vtb + EQ;

    // smem: k1 = max(staging 20480, epilogue 8*4096) = 32768
    //       k2 = 4 * (64*64*4 + 64*64*2)             = 98304
    //       k3 = max(staging 20480, epilogue 8*8192) = 65536
    qkv_gemm_kernel<<<dim3(BWIN, 6), 256, 32768, stream>>>(
        x, w_qkv, b_qkv, Qb, Kb, Vtb);
    attn_kernel<<<dim3(BWIN, 4), 128, 98304, stream>>>(
        Qb, Kb, Vtb, bias_table, rel_index, mask, Yb);
    proj_gemm_kernel<<<dim3(BWIN, 2), 256, 65536, stream>>>(
        Yb, w_proj, b_proj, out);
}
